// LinearAttention_59193239273702
// MI455X (gfx1250) — compile-verified, matrix operand order checked by reading, not run on hardware
//
#include <hip/hip_runtime.h>

// ---------------------------------------------------------------------------
// Linear attention for MI455X (gfx1250): all GEMM stages on v_wmma_f32_16x16x32_bf16,
// weight strips staged to LDS via global_load_async_to_lds_b128 (ASYNCcnt path).
// ---------------------------------------------------------------------------

#define B_DIM 4
#define S_DIM 4096
#define D_DIM 1024
#define H_DIM 16
#define HD    64
#define M_DIM (B_DIM * S_DIM)   // 16384

typedef __attribute__((ext_vector_type(16))) __bf16 v16bf;
typedef __attribute__((ext_vector_type(8)))  float  v8f;
typedef __attribute__((ext_vector_type(4)))  unsigned int u32x4;
typedef __attribute__((ext_vector_type(4)))  int          i32x4;

union FragU {
    v16bf v;
    u32x4 u[2];
    __bf16 h[16];
};

__device__ __forceinline__ v8f zero8() {
    v8f z = {0.f, 0.f, 0.f, 0.f, 0.f, 0.f, 0.f, 0.f};
    return z;
}

// Contiguous fragment load: per-lane two 16B chunks (K 0..7 & 16..23 for lanes
// 0-15, K 8..15 & 24..31 for lanes 16-31 — caller bakes the +8 into the ptr).
__device__ __forceinline__ v16bf load_frag_contig(const __bf16* p) {
    FragU f;
    f.u[0] = *reinterpret_cast<const u32x4*>(p);
    f.u[1] = *reinterpret_cast<const u32x4*>(p + 16);
    return f.v;
}

__device__ __forceinline__ v8f wmma_bf16(v16bf a, v16bf b, v8f c) {
    return __builtin_amdgcn_wmma_f32_16x16x32_bf16(
        /*neg_a=*/false, a, /*neg_b=*/false, b,
        /*c_mod=*/(short)0, c, /*reuse_a=*/false, /*reuse_b=*/false);
}

// ---------------------------------------------------------------------------
// fp32 -> bf16 conversion (grid-stride)
// ---------------------------------------------------------------------------
__global__ __launch_bounds__(256) void la_cvt(const float* __restrict__ src,
                                              __bf16* __restrict__ dst, int n) {
    int i = blockIdx.x * 256 + threadIdx.x;
    const int stride = gridDim.x * 256;
    for (; i < n; i += stride) dst[i] = (__bf16)src[i];
}

// ---------------------------------------------------------------------------
// NT GEMM: out[m,n] = sum_k A[m,k] * W[n,k]
// Block: 8 waves x 32(M) rows each, one shared 64(N) strip of W staged in LDS
// (128 KB, async copy -> ASYNCcnt). Wave: 2 A-fragments x 4 B-fragments ->
// 8 WMMAs per K-step of 32; all 4 B fragments preloaded so DS latency overlaps.
// mode: 0 = bf16 store, 1 = elu+1 then bf16 store, 2 = fp32 store
// ---------------------------------------------------------------------------
__global__ __launch_bounds__(256) void la_gemm_nt(const __bf16* __restrict__ A,
                                                  const __bf16* __restrict__ W,
                                                  void* __restrict__ out,
                                                  int K, int N, int mode) {
    extern __shared__ char smem_raw[];                 // 64 rows x K bf16
    __bf16* smem = (__bf16*)smem_raw;

    const int lane   = threadIdx.x & 31;
    const int wave   = threadIdx.x >> 5;
    const int nStrips = N >> 6;
    const int strip  = blockIdx.x % nStrips;
    const int mGroup = blockIdx.x / nStrips;
    const int n0     = strip * 64;
    const int l15    = lane & 15;
    const int kbase  = (lane >> 4) << 3;   // 0 or 8

    // ---- stage the 64-row weight strip for this block into LDS ----
    {
        const char* gsrc = (const char*)(W + (size_t)n0 * K);   // 64*K*2 bytes, contiguous
        char* ldst = smem_raw;
        const int totalChunks = (64 * K * 2) / 16;              // 16B chunks
#if __has_builtin(__builtin_amdgcn_global_load_async_to_lds_b128)
        for (int c = threadIdx.x; c < totalChunks; c += 256) {
            const int off = c * 16;
            __builtin_amdgcn_global_load_async_to_lds_b128(
                (__attribute__((address_space(1))) i32x4*)(gsrc + off),
                (__attribute__((address_space(3))) i32x4*)(ldst + off),
                /*offset=*/0, /*cpol=*/0);
        }
#if __has_builtin(__builtin_amdgcn_s_wait_asynccnt)
        __builtin_amdgcn_s_wait_asynccnt(0);
#else
        asm volatile("s_wait_asynccnt 0" ::: "memory");
#endif
#else
        for (int c = threadIdx.x; c < totalChunks; c += 256) {
            const int off = c * 16;
            *(u32x4*)(ldst + off) = *(const u32x4*)(gsrc + off);
        }
#endif
        __syncthreads();
    }

    // ---- per-wave 32x64 tile ----
    const int rowbase = mGroup * 256 + wave * 32;
    const __bf16* arow0 = A + (size_t)(rowbase + l15) * K + kbase;
    const __bf16* arow1 = A + (size_t)(rowbase + 16 + l15) * K + kbase;

    const __bf16* wl[4];                               // LDS fragment pointers
#pragma unroll
    for (int t = 0; t < 4; ++t)
        wl[t] = smem + (size_t)(t * 16 + l15) * K + kbase;

    v8f acc0[4], acc1[4];
#pragma unroll
    for (int t = 0; t < 4; ++t) { acc0[t] = zero8(); acc1[t] = zero8(); }

    for (int k0 = 0; k0 < K; k0 += 32) {
        if (k0 + 128 < K) {
            __builtin_prefetch(arow0 + k0 + 128, 0, 1);
            __builtin_prefetch(arow1 + k0 + 128, 0, 1);
        }
        v16bf a0 = load_frag_contig(arow0 + k0);
        v16bf a1 = load_frag_contig(arow1 + k0);
        v16bf bfr[4];                                  // all DS loads in flight
#pragma unroll
        for (int t = 0; t < 4; ++t) bfr[t] = load_frag_contig(wl[t] + k0);
#pragma unroll
        for (int t = 0; t < 4; ++t) {
            acc0[t] = wmma_bf16(a0, bfr[t], acc0[t]);
            acc1[t] = wmma_bf16(a1, bfr[t], acc1[t]);
        }
    }

    // C/D layout: lane L, vgpr r -> row = r + 8*(L>=16), col = L&15 (per tile)
    const int half8 = (lane >> 4) << 3;
#pragma unroll
    for (int sub = 0; sub < 2; ++sub) {
        const int mbase = rowbase + sub * 16 + half8;
        v8f* acc = sub ? acc1 : acc0;
#pragma unroll
        for (int t = 0; t < 4; ++t) {
#pragma unroll
            for (int r = 0; r < 8; ++r) {
                float vl = acc[t][r];
                if (mode == 1) vl = (vl > 0.f) ? (vl + 1.f) : __expf(vl);  // elu(x)+1
                const size_t idx = (size_t)(mbase + r) * N + (n0 + t * 16 + l15);
                if (mode == 2) ((float*)out)[idx] = vl;
                else           ((__bf16*)out)[idx] = (__bf16)vl;
            }
        }
    }
}

// ---------------------------------------------------------------------------
// k_sum[b,h,d] = sum_s k[b,s,h,d]   (fp32 accumulation, global atomics)
// ---------------------------------------------------------------------------
__global__ __launch_bounds__(64) void la_ksum(const __bf16* __restrict__ kb,
                                              float* __restrict__ ksum) {
    const int bh    = blockIdx.x >> 3;
    const int chunk = blockIdx.x & 7;
    const int b = bh >> 4, h = bh & 15;
    const int d = threadIdx.x;
    const __bf16* p = kb + ((size_t)(b * S_DIM + chunk * (S_DIM / 8))) * D_DIM + h * HD + d;
    float s = 0.f;
    for (int i = 0; i < S_DIM / 8; ++i) s += (float)p[(size_t)i * D_DIM];
    atomicAdd(&ksum[bh * HD + d], s);
}

// ---------------------------------------------------------------------------
// KV^T[b,h,e,d] = sum_s v[b,s,h,e] * k[b,s,h,d]   (stored transposed, bf16)
// One block per (b,h); 8 waves each sum a 512-row slice of S into 4x4 WMMA
// accumulators; cross-wave reduce via LDS ds_add_f32 atomics.
// ---------------------------------------------------------------------------
__global__ __launch_bounds__(256) void la_kv(const __bf16* __restrict__ kb,
                                             const __bf16* __restrict__ vb,
                                             __bf16* __restrict__ kvt) {
    __shared__ float red[HD * HD];
    const int bh = blockIdx.x;
    const int b = bh >> 4, h = bh & 15;

    for (int i = threadIdx.x; i < HD * HD; i += 256) red[i] = 0.f;
    __syncthreads();

    const int lane = threadIdx.x & 31;
    const int wave = threadIdx.x >> 5;
    const int l15  = lane & 15;
    const int shalf = (lane >> 4) << 3;     // +8 K-offset for upper half-wave
    const size_t base = (size_t)b * S_DIM * D_DIM + h * HD;

    v8f acc[4][4];
#pragma unroll
    for (int et = 0; et < 4; ++et)
#pragma unroll
        for (int dt = 0; dt < 4; ++dt) acc[et][dt] = zero8();

    const int sBeg = wave * (S_DIM / 8);
    for (int s0 = sBeg; s0 < sBeg + S_DIM / 8; s0 += 32) {
        // B fragments from k (K = s, N = d): gathered, coalesced across lanes
        v16bf fb[4];
#pragma unroll
        for (int dt = 0; dt < 4; ++dt) {
            FragU f;
#pragma unroll
            for (int j = 0; j < 16; ++j) {
                const int s = s0 + shalf + (j & 7) + ((j >> 3) << 4);
                f.h[j] = kb[base + (size_t)s * D_DIM + dt * 16 + l15];
            }
            fb[dt] = f.v;
        }
        // A fragments from v^T (M = e, K = s)
#pragma unroll
        for (int et = 0; et < 4; ++et) {
            FragU f;
#pragma unroll
            for (int j = 0; j < 16; ++j) {
                const int s = s0 + shalf + (j & 7) + ((j >> 3) << 4);
                f.h[j] = vb[base + (size_t)s * D_DIM + et * 16 + l15];
            }
            v16bf fa = f.v;
#pragma unroll
            for (int dt = 0; dt < 4; ++dt)
                acc[et][dt] = wmma_bf16(fa, fb[dt], acc[et][dt]);
        }
    }

    const int rbase = (lane >> 4) << 3;
#pragma unroll
    for (int et = 0; et < 4; ++et)
#pragma unroll
        for (int dt = 0; dt < 4; ++dt)
#pragma unroll
            for (int r = 0; r < 8; ++r)
                atomicAdd(&red[(et * 16 + rbase + r) * HD + dt * 16 + l15],
                          acc[et][dt][r]);
    __syncthreads();

    for (int i = threadIdx.x; i < HD * HD; i += 256)
        kvt[(size_t)bh * (HD * HD) + i] = (__bf16)red[i];
}

// ---------------------------------------------------------------------------
// out[b,s,h,e] = (q[b,s,h,:] @ KV[b,h,:,e]) / (q·k_sum + eps), stored bf16.
// One wave per 16(s) x 64(e) tile; K = 64 so just two WMMA K-steps.
// ---------------------------------------------------------------------------
__global__ __launch_bounds__(256) void la_attn(const __bf16* __restrict__ qb,
                                               const __bf16* __restrict__ kvt,
                                               const float* __restrict__ ksum,
                                               __bf16* __restrict__ attn) {
    const int lane = threadIdx.x & 31;
    const int wave = threadIdx.x >> 5;
    const int id = blockIdx.x * 8 + wave;
    const int bh = id >> 8;           // 256 s-tiles per (b,h)
    const int sTile = id & 255;
    const int b = bh >> 4, h = bh & 15;
    const int l15 = lane & 15;
    const int dbase = (lane >> 4) << 3;

    const size_t qrow = ((size_t)(b * S_DIM + sTile * 16 + l15)) * D_DIM + h * HD + dbase;
    const __bf16* kvbase = kvt + (size_t)bh * (HD * HD);   // [e*64 + d]

    v8f acc[4];
#pragma unroll
    for (int t = 0; t < 4; ++t) acc[t] = zero8();

#pragma unroll
    for (int k0 = 0; k0 < HD; k0 += 32) {
        v16bf a = load_frag_contig(qb + qrow + k0);
        v16bf bfr[4];
#pragma unroll
        for (int t = 0; t < 4; ++t)
            bfr[t] = load_frag_contig(kvbase + (size_t)(t * 16 + l15) * HD + k0 + dbase);
#pragma unroll
        for (int t = 0; t < 4; ++t)
            acc[t] = wmma_bf16(a, bfr[t], acc[t]);
    }

    // z for row m = lane&15 (both half-waves compute; shuffle broadcasts)
    const __bf16* qz = qb + ((size_t)(b * S_DIM + sTile * 16 + l15)) * D_DIM + h * HD;
    const float* ks = ksum + bh * HD;
    float z = 0.f;
#pragma unroll 8
    for (int d = 0; d < HD; ++d) z += (float)qz[d] * ks[d];

    float zr[8];
#pragma unroll
    for (int r = 0; r < 8; ++r) zr[r] = __shfl(z, r + ((lane >> 4) << 3), 32);

    const int mbase = sTile * 16 + ((lane >> 4) << 3);
#pragma unroll
    for (int t = 0; t < 4; ++t)
#pragma unroll
        for (int r = 0; r < 8; ++r) {
            const float o = acc[t][r] / (zr[r] + 1e-6f);
            attn[((size_t)(b * S_DIM + mbase + r)) * D_DIM + h * HD + t * 16 + l15] =
                (__bf16)o;
        }
}

// ---------------------------------------------------------------------------
// Launcher
// ---------------------------------------------------------------------------
extern "C" void kernel_launch(void* const* d_in, const int* in_sizes, int n_in,
                              void* d_out, int out_size, void* d_ws, size_t ws_size,
                              hipStream_t stream) {
    const float* x  = (const float*)d_in[0];
    const float* Wq = (const float*)d_in[1];
    const float* Wk = (const float*)d_in[2];
    const float* Wv = (const float*)d_in[3];
    const float* Wo = (const float*)d_in[4];

    char* ws = (char*)d_ws;
    const size_t MB = 1024ull * 1024ull;
    __bf16* xb   = (__bf16*)(ws);              // 32 MiB (reused as attn output)
    __bf16* wqb  = (__bf16*)(ws + 32 * MB);    // 2 MiB each
    __bf16* wkb  = (__bf16*)(ws + 34 * MB);
    __bf16* wvb  = (__bf16*)(ws + 36 * MB);
    __bf16* wob  = (__bf16*)(ws + 38 * MB);
    __bf16* qb   = (__bf16*)(ws + 40 * MB);    // 32 MiB
    __bf16* kb   = (__bf16*)(ws + 72 * MB);    // 32 MiB
    __bf16* vb   = (__bf16*)(ws + 104 * MB);   // 32 MiB
    __bf16* kvt  = (__bf16*)(ws + 136 * MB);   // 512 KiB
    float*  ksum = (float*)(ws + 136 * MB + 512 * 1024);  // 16 KiB

    const int nElemX = M_DIM * D_DIM;
    const int nElemW = D_DIM * D_DIM;

    la_cvt<<<2048, 256, 0, stream>>>(x, xb, nElemX);
    la_cvt<<<512, 256, 0, stream>>>(Wq, wqb, nElemW);
    la_cvt<<<512, 256, 0, stream>>>(Wk, wkb, nElemW);
    la_cvt<<<512, 256, 0, stream>>>(Wv, wvb, nElemW);
    la_cvt<<<512, 256, 0, stream>>>(Wo, wob, nElemW);

    const int gemmBlocks = (M_DIM / 256) * (D_DIM / 64);   // 64 * 16 = 1024
    const size_t ldsBytes = 64 * D_DIM * sizeof(__bf16);   // 128 KiB weight strip

    la_gemm_nt<<<gemmBlocks, 256, ldsBytes, stream>>>(xb, wqb, qb, D_DIM, D_DIM, 1);
    la_gemm_nt<<<gemmBlocks, 256, ldsBytes, stream>>>(xb, wkb, kb, D_DIM, D_DIM, 1);
    la_gemm_nt<<<gemmBlocks, 256, ldsBytes, stream>>>(xb, wvb, vb, D_DIM, D_DIM, 0);

    (void)hipMemsetAsync(ksum, 0, B_DIM * H_DIM * HD * sizeof(float), stream);
    la_ksum<<<B_DIM * H_DIM * 8, 64, 0, stream>>>(kb, ksum);
    la_kv<<<B_DIM * H_DIM, 256, 0, stream>>>(kb, vb, kvt);

    la_attn<<<2048, 256, 0, stream>>>(qb, kvt, ksum, xb /* reuse xb as attn */);

    la_gemm_nt<<<gemmBlocks, 256, ldsBytes, stream>>>(xb, wob, d_out, D_DIM, D_DIM, 2);
}
